// GraphLaplacianPPO_19885698580850
// MI455X (gfx1250) — matrix operand order repaired and checked by reading, not compiled, not run on hardware
//
#include <hip/hip_runtime.h>

// ---------------------------------------------------------------------------
// GraphLaplacianPPO forward for MI455X (gfx1250, wave32, WMMA f16->f32)
//   obs[B,17] -> tanh fc 256 -> tanh fc 256 = feat
//   route: argmin_m || (feat-mean)*rsqrt(var+eps) - centers[m] ||^2
//   mu = feat @ head_muW[idx] + head_mub[idx]; log_std = clamp(head_logstd[idx])
//   v: feat -> tanh fc 256 -> tanh fc 256 -> fc 1
//   out[B,13] = [mu(6), log_std(6), v(1)]
// ---------------------------------------------------------------------------

typedef __attribute__((ext_vector_type(16))) _Float16 v16h;
typedef __attribute__((ext_vector_type(8)))  float    v8f;

#define HIDN 256
#define OBSN 17
#define ACTN 6
#define MCH  8
#define OUTC 13

// packed f16 weight fragment offsets inside d_ws (units: _Float16 elements)
// W1: 16 tiles * 512, W2/V1/V2: 128 tiles * 512 each
#define W1_OFF 0
#define W2_OFF 8192
#define V1_OFF 73728
#define V2_OFF 139264
// total = 204800 halves = 409600 bytes of workspace

union Frag16 { v16h v; uint4 q[2]; };

__device__ __forceinline__ v16h loadB_global(const _Float16* p) {
  Frag16 f;
  const uint4* q = (const uint4*)p;
  f.q[0] = q[0];
  f.q[1] = q[1];
  return f.v;
}

// A fragment (16-bit, 16x32): lane l holds row m=l&15; element e -> k =
// kbase + e for e<8, kbase + 16 + (e-8) for e>=8, with kbase = kt*32 + 8*(l>>4)
__device__ __forceinline__ v16h loadA_lds(const _Float16* row, int kbase) {
  Frag16 f;
  f.q[0] = *(const uint4*)(row + kbase);
  f.q[1] = *(const uint4*)(row + kbase + 16);
  return f.v;
}

// Branch-free tanh: tanh(x) = 1 - 2/(exp2(2*log2e*x)+1), clamped to +-9 so the
// exp2 never overflows.  One v_exp_f32 (TRANS, co-executes with XDL WMMA) +
// one v_rcp_f32 + 4 VALU ops; accuracy ~1e-7 absolute, far beyond the f16
// storage precision of the activations.
__device__ __forceinline__ float fast_tanh(float x) {
  const float xc = fminf(9.0f, fmaxf(-9.0f, x));
  const float t  = __builtin_amdgcn_exp2f(xc * 2.8853900817779268f);
  return 1.0f - 2.0f * __builtin_amdgcn_rcpf(t + 1.0f);
}

// One 16x256 tanh-FC layer with K=256 (8 WMMA k-steps per 16-col tile).
// src/dst are this wave's private [16][256] LDS buffers.
__device__ __forceinline__ void fc256_tanh(const _Float16 (*src)[HIDN],
                                           _Float16 (*dst)[HIDN],
                                           const _Float16* wp,
                                           const float* bias,
                                           int lane) {
  const int col = lane & 15;
  const int g   = lane >> 4;
  const _Float16* arow = src[col];
  #pragma unroll
  for (int nt = 0; nt < 16; ++nt) {
    const float b = bias[nt * 16 + col];
    v8f acc = {b, b, b, b, b, b, b, b};
    #pragma unroll
    for (int kt = 0; kt < 8; ++kt) {
      v16h a = loadA_lds(arow, kt * 32 + 8 * g);
      v16h w = loadB_global(wp + (nt * 8 + kt) * 512 + lane * 16);
      acc = __builtin_amdgcn_wmma_f32_16x16x32_f16(false, a, false, w,
                                                   (short)0, acc, false, false);
    }
    // D layout: vgpr r -> row m = r + 8*g, col n = nt*16 + (lane&15)
    #pragma unroll
    for (int r = 0; r < 8; ++r) {
      dst[r + 8 * g][nt * 16 + col] = (_Float16)fast_tanh(acc[r]);
    }
  }
}

// ---------------------------------------------------------------------------
// Prep: pack f32 weight matrices into f16 B-fragment-linear layout.
// B fragment layout (16-bit, 32x16): lane l holds col n=l&15; element e -> k =
// ktile*32 + 8*(l>>4) + e for e<8, + 16 + (e-8) for e>=8.  One block (32 threads) packs one tile (512 halves).
// ---------------------------------------------------------------------------
__global__ void pack_weights_kernel(const float* __restrict__ W1,
                                    const float* __restrict__ W2,
                                    const float* __restrict__ V1,
                                    const float* __restrict__ V2,
                                    _Float16* __restrict__ ws) {
  const int b   = blockIdx.x;       // 0..399
  const int l   = threadIdx.x;      // 0..31
  const int col = l & 15;
  const int g   = l >> 4;

  if (b < 16) {
    // enc_W1 [17,256], K zero-padded to 32; one k-tile per n-tile
    const int nt = b;
    _Float16* dst = ws + W1_OFF + nt * 512 + l * 16;
    const int n = nt * 16 + col;
    #pragma unroll
    for (int e = 0; e < 16; ++e) {
      const int k = ((e < 8) ? e : (e + 8)) + 8 * g;
      dst[e] = (k < OBSN) ? (_Float16)W1[k * HIDN + n] : (_Float16)0.0f;
    }
  } else {
    const float* W;
    int t, off;
    if (b < 144)      { W = W2; t = b - 16;  off = W2_OFF; }
    else if (b < 272) { W = V1; t = b - 144; off = V1_OFF; }
    else              { W = V2; t = b - 272; off = V2_OFF; }
    const int nt = t >> 3;
    const int kt = t & 7;
    _Float16* dst = ws + off + t * 512 + l * 16;
    const int n = nt * 16 + col;
    #pragma unroll
    for (int e = 0; e < 16; ++e) {
      const int k = kt * 32 + ((e < 8) ? e : (e + 8)) + 8 * g;
      dst[e] = (_Float16)W[k * HIDN + n];
    }
  }
}

// ---------------------------------------------------------------------------
// Main fused kernel: 4 waves per block, 16 rows per wave, 64 rows per block.
// ---------------------------------------------------------------------------
__global__ __launch_bounds__(128)
void ppo_forward_kernel(const float* __restrict__ obs,
                        const float* __restrict__ enc_b1,
                        const float* __restrict__ enc_b2,
                        const float* __restrict__ head_muW,
                        const float* __restrict__ head_mub,
                        const float* __restrict__ head_logstd,
                        const float* __restrict__ val_b1,
                        const float* __restrict__ val_b2,
                        const float* __restrict__ val_W3,
                        const float* __restrict__ val_b3,
                        const float* __restrict__ centers,
                        const float* __restrict__ stats_mean,
                        const float* __restrict__ stats_var,
                        const _Float16* __restrict__ wpack,
                        float* __restrict__ out) {
  __shared__ _Float16 sAct[4][2][16][HIDN];   // 64 KB: ping-pong per wave

  const int lane = threadIdx.x & 31;
  const int wave = threadIdx.x >> 5;
  const int col  = lane & 15;
  const int g    = lane >> 4;
  const int rowBase = blockIdx.x * 64 + wave * 16;

  _Float16 (*buf0)[HIDN] = sAct[wave][0];
  _Float16 (*buf1)[HIDN] = sAct[wave][1];

  // ---- layer 1: obs (K padded 17->32) -> buf0 -------------------------------
  {
    v16h a1;
    const float* orow = obs + (size_t)(rowBase + col) * OBSN;   // A row m = lane&15
    #pragma unroll
    for (int e = 0; e < 16; ++e) {
      const int k = ((e < 8) ? e : (e + 8)) + 8 * g;            // A-frag k pattern
      a1[e] = (k < OBSN) ? (_Float16)orow[k] : (_Float16)0.0f;
    }
    #pragma unroll
    for (int nt = 0; nt < 16; ++nt) {
      const float b = enc_b1[nt * 16 + col];
      v8f acc = {b, b, b, b, b, b, b, b};
      v16h w = loadB_global(wpack + W1_OFF + nt * 512 + lane * 16);
      acc = __builtin_amdgcn_wmma_f32_16x16x32_f16(false, a1, false, w,
                                                   (short)0, acc, false, false);
      #pragma unroll
      for (int r = 0; r < 8; ++r) {
        buf0[r + 8 * g][nt * 16 + col] = (_Float16)fast_tanh(acc[r]);
      }
    }
  }
  __syncthreads();

  // ---- layer 2: buf0 -> feat (buf1) ----------------------------------------
  fc256_tanh(buf0, buf1, wpack + W2_OFF, enc_b2, lane);
  __syncthreads();

  // ---- scalar phase: routing + mu head + log_std (one lane per row) --------
  if (lane < 16) {
    const int grow = rowBase + lane;
    const _Float16* f = buf1[lane];

    float d2[MCH];
    #pragma unroll
    for (int m = 0; m < MCH; ++m) d2[m] = 0.0f;
    for (int d = 0; d < HIDN; ++d) {
      const float fv = (float)f[d];
      const float zw = (fv - stats_mean[d]) * rsqrtf(stats_var[d] + 1e-6f);
      #pragma unroll
      for (int m = 0; m < MCH; ++m) {
        const float df = zw - centers[m * HIDN + d];
        d2[m] += df * df;
      }
    }
    int idx = 0;
    float best = d2[0];
    #pragma unroll
    for (int m = 1; m < MCH; ++m) {
      if (d2[m] < best) { best = d2[m]; idx = m; }
    }

    float mu[ACTN];
    #pragma unroll
    for (int a = 0; a < ACTN; ++a) mu[a] = head_mub[idx * ACTN + a];
    for (int d = 0; d < HIDN; ++d) {
      const float fv = (float)f[d];
      const float* W = head_muW + ((size_t)idx * HIDN + d) * ACTN;
      #pragma unroll
      for (int a = 0; a < ACTN; ++a) mu[a] += fv * W[a];
    }

    float* o = out + (size_t)grow * OUTC;
    #pragma unroll
    for (int a = 0; a < ACTN; ++a) o[a] = mu[a];
    #pragma unroll
    for (int a = 0; a < ACTN; ++a) {
      float ls = head_logstd[idx * ACTN + a];
      ls = fminf(2.0f, fmaxf(-20.0f, ls));
      o[ACTN + a] = ls;
    }
  }
  __syncthreads();

  // ---- value head: feat(buf1) -> buf0 -> buf1 ------------------------------
  fc256_tanh(buf1, buf0, wpack + V1_OFF, val_b1, lane);
  __syncthreads();
  fc256_tanh(buf0, buf1, wpack + V2_OFF, val_b2, lane);
  __syncthreads();

  if (lane < 16) {
    const _Float16* f = buf1[lane];
    float s = val_b3[0];
    for (int d = 0; d < HIDN; ++d) s += (float)f[d] * val_W3[d];
    out[(size_t)(rowBase + lane) * OUTC + 12] = s;
  }
}

// ---------------------------------------------------------------------------
extern "C" void kernel_launch(void* const* d_in, const int* in_sizes, int n_in,
                              void* d_out, int out_size, void* d_ws, size_t ws_size,
                              hipStream_t stream) {
  (void)in_sizes; (void)n_in; (void)out_size; (void)ws_size;

  const float* obs         = (const float*)d_in[0];
  const float* enc_W1      = (const float*)d_in[1];
  const float* enc_b1      = (const float*)d_in[2];
  const float* enc_W2      = (const float*)d_in[3];
  const float* enc_b2      = (const float*)d_in[4];
  const float* head_muW    = (const float*)d_in[5];
  const float* head_mub    = (const float*)d_in[6];
  const float* head_logstd = (const float*)d_in[7];
  const float* val_W1      = (const float*)d_in[8];
  const float* val_b1      = (const float*)d_in[9];
  const float* val_W2      = (const float*)d_in[10];
  const float* val_b2      = (const float*)d_in[11];
  const float* val_W3      = (const float*)d_in[12];
  const float* val_b3      = (const float*)d_in[13];
  const float* centers     = (const float*)d_in[14];
  const float* stats_mean  = (const float*)d_in[15];
  const float* stats_var   = (const float*)d_in[16];

  _Float16* ws = (_Float16*)d_ws;
  float* out   = (float*)d_out;

  // 1) pack weights to f16 WMMA B-fragments (16 + 3*128 = 400 tiles)
  pack_weights_kernel<<<400, 32, 0, stream>>>(enc_W1, enc_W2, val_W1, val_W2, ws);

  // 2) fused forward: 131072 rows / 64 rows-per-block = 2048 blocks
  ppo_forward_kernel<<<2048, 128, 0, stream>>>(
      obs, enc_b1, enc_b2, head_muW, head_mub, head_logstd,
      val_b1, val_b2, val_W3, val_b3, centers, stats_mean, stats_var,
      ws, out);
}
